// RealBDH_18485539242697
// MI455X (gfx1250) — matrix operand order repaired and checked by reading, not compiled, not run on hardware
//
#include <hip/hip_runtime.h>

typedef __attribute__((ext_vector_type(16))) __bf16 v16bf;
typedef __attribute__((ext_vector_type(8)))  __bf16 v8bf;
typedef __attribute__((ext_vector_type(8)))  float  v8f;
typedef __attribute__((ext_vector_type(16))) float  v16f;

#define NDIM 2048
#define BROWS 64
#define HIDD 256

// float -> bf16, round-to-nearest-even (for scalar paths)
__device__ __forceinline__ unsigned short f2bf(float f) {
    unsigned u = __builtin_bit_cast(unsigned, f);
    unsigned rounding = 0x7FFFu + ((u >> 16) & 1u);
    return (unsigned short)((u + rounding) >> 16);
}
__device__ __forceinline__ unsigned pack2(float a, float b) {
    return (unsigned)f2bf(a) | ((unsigned)f2bf(b) << 16);
}

// ---------------------------------------------------------------------------
// Gather last-token embeddings, relu, emit f32 + bf16 copies of acts0.
// grid = 64 blocks (one per batch row), block = 256.
__global__ __launch_bounds__(256)
void gather_relu_kernel(const int* __restrict__ idx, int T,
                        const float* __restrict__ token_embed,
                        float* __restrict__ acts0_f,
                        unsigned* __restrict__ acts0_b /* packed bf16 pairs */) {
    const int b = blockIdx.x;
    const int tok = idx[(size_t)b * T + (T - 1)];
    const float* src = token_embed + (size_t)tok * NDIM;
    for (int j = threadIdx.x * 2; j < NDIM; j += 2 * blockDim.x) {
        float2 v = *(const float2*)(src + j);
        v.x = fmaxf(v.x, 0.0f);
        v.y = fmaxf(v.y, 0.0f);
        *(float2*)(acts0_f + (size_t)b * NDIM + j) = v;
        acts0_b[((size_t)b * NDIM + j) >> 1] = pack2(v.x, v.y);
    }
}

// a2 = bf16(acts0 + 0.5 * propagated)
__global__ __launch_bounds__(256)
void a2_kernel(const float* __restrict__ acts0, const float* __restrict__ prop,
               unsigned* __restrict__ o, int n) {
    int j = (blockIdx.x * blockDim.x + threadIdx.x) * 2;
    if (j + 1 < n) {
        float2 av = *(const float2*)(acts0 + j);
        float2 pv = *(const float2*)(prop + j);
        o[j >> 1] = pack2(av.x + 0.5f * pv.x, av.y + 0.5f * pv.y);
    }
}

// ---------------------------------------------------------------------------
// Generic 64xNN = (64xK bf16) x (NNxK f32)^T GEMM with the B operand
// converted f32 -> bf16 on the fly (and optionally formed as an elementwise
// product of two f32 matrices, for adjacency * synapse_state). Accumulation
// via v_wmma_f32_16x16x32_bf16 in f32. Each big f32 matrix is read from HBM
// exactly once; no bf16 copy is ever materialized.
//
// Block = 128 threads = 4 waves; wave w computes rows [16w, 16w+16);
// blockIdx.x selects the 16-wide N tile.
//
// Fragment layouts (A/B: contiguous k order chosen here; C/D per ISA; wave32, hi = lane>>4, lr = lane&15):
//   A (16x32 bf16): lane covers row lr; VGPRs 0-7 = K {hi*16..hi*16+15},
//                   two 16B contiguous loads.
//   B (32x16 bf16): lane covers col lr; VGPRs 0-7 = K {hi*16..hi*16+15},
//                   one contiguous run -> one 64B f32 load, cvt to bf16.
//   C/D (16x16 f32): VGPR r, lane -> element [r + hi*8][lr].
template <bool MULB>
__global__ __launch_bounds__(128)
void gemm_fusedB_wmma(const __bf16* __restrict__ A,  // 64 x K bf16, row-major
                      const float* __restrict__ B0,  // NN x K f32, row-major
                      const float* __restrict__ B1,  // NN x K f32 or unused
                      float* __restrict__ C,         // 64 x NN f32
                      int K, int NN,
                      const float* __restrict__ bias, int relu) {
    const int wave = threadIdx.x >> 5;
    const int lane = threadIdx.x & 31;
    const int hi   = lane >> 4;    // 0: lanes 0-15, 1: lanes 16-31
    const int lr   = lane & 15;
    const int m0   = wave * 16;
    const int n0   = blockIdx.x * 16;

    const __bf16* Ar  = A + (size_t)(m0 + lr) * K;   // this lane's A row
    const float*  B0r = B0 + (size_t)(n0 + lr) * K;  // this lane's B column
    const float*  B1r = MULB ? (B1 + (size_t)(n0 + lr) * K) : nullptr;

    v8f acc = {};
    for (int k0 = 0; k0 < K; k0 += 32) {
        // A fragment: two contiguous 16B bf16 loads
        v8bf alo = *(const v8bf*)(Ar + k0 + hi * 16);
        v8bf ahi = *(const v8bf*)(Ar + k0 + hi * 16 + 8);
        v16bf a = __builtin_shufflevector(alo, ahi,
                                          0, 1, 2, 3, 4, 5, 6, 7,
                                          8, 9, 10, 11, 12, 13, 14, 15);
        // B fragment: 16 consecutive f32 (64B), optional product, cvt->bf16
        v16f x = *(const v16f*)(B0r + k0 + hi * 16);
        if (MULB) {
            v16f y = *(const v16f*)(B1r + k0 + hi * 16);
            x = x * y;
        }
        v16bf b = __builtin_convertvector(x, v16bf);
        acc = __builtin_amdgcn_wmma_f32_16x16x32_bf16(
            /*neg_a=*/false, a, /*neg_b=*/false, b,
            /*c_mod=*/(short)0, acc, /*reuse_a=*/false, /*reuse_b=*/false);
    }
    // C/D layout: VGPR r, lane -> element [r + hi*8][lr]
    const int n = n0 + lr;
    const float bv = bias ? bias[n] : 0.0f;
#pragma unroll
    for (int r = 0; r < 8; ++r) {
        const int m = m0 + r + hi * 8;
        float v = acc[r] + bv;
        if (relu) v = fmaxf(v, 0.0f);
        C[(size_t)m * NN + n] = v;
    }
}

// ---------------------------------------------------------------------------
// LayerNorm per row (population variance, matching jnp.var), output bf16.
// grid = 64 rows, block = 256 (8 waves, wave32 shuffles).
__global__ __launch_bounds__(256)
void ln_kernel(const float* __restrict__ x_all,
               const float* __restrict__ g, const float* __restrict__ bb,
               unsigned short* __restrict__ out, float eps) {
    const int row = blockIdx.x;
    const float* x = x_all + (size_t)row * NDIM;
    float s = 0.0f, s2 = 0.0f;
    for (int i = threadIdx.x; i < NDIM; i += blockDim.x) {
        float v = x[i];
        s += v;
        s2 += v * v;
    }
    for (int off = 16; off > 0; off >>= 1) {
        s  += __shfl_down(s, off);
        s2 += __shfl_down(s2, off);
    }
    __shared__ float ss[8], ss2[8];
    const int wave = threadIdx.x >> 5, lane = threadIdx.x & 31;
    if (lane == 0) { ss[wave] = s; ss2[wave] = s2; }
    __syncthreads();
    if (threadIdx.x == 0) {
        float t = 0.0f, t2 = 0.0f;
        for (int w = 0; w < 8; ++w) { t += ss[w]; t2 += ss2[w]; }
        ss[0] = t; ss2[0] = t2;
    }
    __syncthreads();
    const float mean = ss[0] * (1.0f / NDIM);
    const float var  = ss2[0] * (1.0f / NDIM) - mean * mean;
    const float inv  = rsqrtf(var + eps);
    for (int i = threadIdx.x; i < NDIM; i += blockDim.x) {
        float v = (x[i] - mean) * inv * g[i] + bb[i];
        out[(size_t)row * NDIM + i] = f2bf(v);
    }
}

// ---------------------------------------------------------------------------
// Final classifier: 64 rows x 2 labels, K = 256. One thread per (row,label).
__global__ __launch_bounds__(128)
void cls_kernel(const float* __restrict__ hidden,
                const float* __restrict__ W_cls, const float* __restrict__ b_cls,
                float* __restrict__ out) {
    const int t = threadIdx.x;
    const int b = t >> 1;
    const int l = t & 1;
    const float* h = hidden + (size_t)b * HIDD;
    const float* w = W_cls + (size_t)l * HIDD;
    float s = 0.0f;
    for (int k = 0; k < HIDD; ++k) s += h[k] * w[k];
    out[(size_t)b * 2 + l] = s + b_cls[l];
}

// ---------------------------------------------------------------------------
extern "C" void kernel_launch(void* const* d_in, const int* in_sizes, int n_in,
                              void* d_out, int out_size, void* d_ws, size_t ws_size,
                              hipStream_t stream) {
    const int*   idx         = (const int*)d_in[0];
    const float* token_embed = (const float*)d_in[1];
    const float* adjacency   = (const float*)d_in[2];
    const float* synapse     = (const float*)d_in[3];
    const float* W_local     = (const float*)d_in[4];
    const float* ln_g        = (const float*)d_in[5];
    const float* ln_b        = (const float*)d_in[6];
    const float* W_read      = (const float*)d_in[7];
    const float* b_read      = (const float*)d_in[8];
    const float* W_cls       = (const float*)d_in[9];
    const float* b_cls       = (const float*)d_in[10];
    float* out = (float*)d_out;

    const int T = in_sizes[0] / BROWS;  // 512

    // Workspace partition (bytes); total ~2.4 MB
    char* ws = (char*)d_ws;
    size_t off = 0;
    float*  acts0_f  = (float*)(ws + off);  off += (size_t)BROWS * NDIM * 4;  // 512 KB
    __bf16* acts0_b  = (__bf16*)(ws + off); off += (size_t)BROWS * NDIM * 2;  // 256 KB
    float*  prop_f   = (float*)(ws + off);  off += (size_t)BROWS * NDIM * 4;  // 512 KB
    __bf16* a2_b     = (__bf16*)(ws + off); off += (size_t)BROWS * NDIM * 2;  // 256 KB
    float*  acts2_f  = (float*)(ws + off);  off += (size_t)BROWS * NDIM * 4;  // 512 KB
    __bf16* pooled_b = (__bf16*)(ws + off); off += (size_t)BROWS * NDIM * 2;  // 256 KB
    float*  hidden_f = (float*)(ws + off);  off += (size_t)BROWS * HIDD * 4;  //  64 KB
    (void)ws_size; (void)n_in; (void)out_size;

    // 1) Gather + relu (f32 + bf16 copies of acts0)
    gather_relu_kernel<<<BROWS, 256, 0, stream>>>(idx, T, token_embed,
                                                  acts0_f, (unsigned*)acts0_b);

    // 2) GEMM1: propagated = acts0 @ (adjacency * synapse)^T
    //    (elementwise product + bf16 conversion fused into the B-fragment load)
    gemm_fusedB_wmma<true><<<NDIM / 16, 128, 0, stream>>>(
        acts0_b, adjacency, synapse, prop_f, NDIM, NDIM, nullptr, 0);

    // 3) a2 = bf16(acts0 + 0.5 * propagated)
    a2_kernel<<<(BROWS * NDIM) / (2 * 256), 256, 0, stream>>>(
        acts0_f, prop_f, (unsigned*)a2_b, BROWS * NDIM);

    // 4) GEMM2: acts2 = relu(a2 @ W_local^T)   (fused f32->bf16 on W_local)
    gemm_fusedB_wmma<false><<<NDIM / 16, 128, 0, stream>>>(
        a2_b, W_local, nullptr, acts2_f, NDIM, NDIM, nullptr, 1);

    // 5) LayerNorm -> bf16 pooled
    ln_kernel<<<BROWS, 256, 0, stream>>>(acts2_f, ln_g, ln_b,
                                         (unsigned short*)pooled_b, 1e-5f);

    // 6) GEMM3: hidden = relu(pooled @ W_read^T + b_read)
    gemm_fusedB_wmma<false><<<HIDD / 16, 128, 0, stream>>>(
        pooled_b, W_read, nullptr, hidden_f, NDIM, HIDD, b_read, 1);

    // 7) Classifier: logits = hidden @ W_cls^T + b_cls
    cls_kernel<<<1, 128, 0, stream>>>(hidden_f, W_cls, b_cls, out);
}
